// Seq2seq_73340861546581
// MI455X (gfx1250) — compile-verified, matrix operand order checked by reading, not run on hardware
//
#include <hip/hip_runtime.h>

// ---------------------------------------------------------------------------
// Seq2seq attentional GRU decoder for MI455X (gfx1250, wave32, WMMA).
// Recurrence runs sequentially (small), the dominant vocab projection (93% of
// FLOPs) is hoisted into one bf16 WMMA GEMM with async-to-LDS B staging.
// bf16 lin_w (98 MB) is L2-resident (192 MB global L2) -> compute bound.
// ---------------------------------------------------------------------------

typedef __bf16 bf16;
typedef __attribute__((ext_vector_type(16))) __bf16 v16bf;
typedef __attribute__((ext_vector_type(8)))  __bf16 v8bf;
typedef __attribute__((ext_vector_type(8)))  float  v8f;

#define NB   64      // batch
#define NS   64      // source length
#define NC   1024    // context dim
#define NH   512     // hidden
#define NE   512     // embed
#define NV   32000   // vocab
#define NT   32      // time steps
#define NK1  1536    // E + C == H + C

// ---- WMMA helpers ---------------------------------------------------------
// 16-bit A/B fragment (16x32 MxK for A; KxN mirrored for B built from
// row-major W[n][k]): lane half h holds K runs {k0+8h .. k0+8h+7} and
// {k0+16+8h .. k0+16+8h+7} -> two 16-byte loads per lane.
__device__ __forceinline__ v16bf ld_frag(const bf16* __restrict__ base,
                                         int row, int ld, int k0, int lane) {
  const bf16* p = base + (size_t)row * (size_t)ld
                       + (size_t)(k0 + ((lane >> 4) << 3));
  v8bf lo = *(const v8bf*)p;
  v8bf hi = *(const v8bf*)(p + 16);
  v16bf r;
#pragma unroll
  for (int i = 0; i < 8; ++i) { r[i] = lo[i]; r[8 + i] = hi[i]; }
  return r;
}

__device__ __forceinline__ v16bf combine(v8bf lo, v8bf hi) {
  v16bf r;
#pragma unroll
  for (int i = 0; i < 8; ++i) { r[i] = lo[i]; r[8 + i] = hi[i]; }
  return r;
}

__device__ __forceinline__ v8f wmma_bf16(v16bf a, v16bf b, v8f c) {
  // (neg_a, A, neg_b, B, c_mod, C, reuse_a, reuse_b)
  return __builtin_amdgcn_wmma_f32_16x16x32_bf16(false, a, false, b,
                                                 (short)0, c, false, false);
}

// ---- phase 0: weight conversion ------------------------------------------
__global__ __launch_bounds__(256) void k_cvt(const float* __restrict__ in,
                                             bf16* __restrict__ out, int n) {
  int i = (blockIdx.x * 256 + threadIdx.x) * 4;
  if (i + 3 < n) {
    float4 v = *(const float4*)(in + i);
    out[i + 0] = (bf16)v.x; out[i + 1] = (bf16)v.y;
    out[i + 2] = (bf16)v.z; out[i + 3] = (bf16)v.w;
  }
}

__global__ __launch_bounds__(256) void k_cvt_t(const float* __restrict__ in,
                                               bf16* __restrict__ out,
                                               int rows, int cols) {
  int idx = blockIdx.x * 256 + threadIdx.x;
  if (idx < rows * cols) {
    int r = idx / cols, c = idx % cols;
    out[(size_t)c * rows + r] = (bf16)in[idx];   // (H,C) -> (C,H) bf16
  }
}

__global__ __launch_bounds__(256) void k_init(const float* __restrict__ h0,
                                              const float* __restrict__ c0,
                                              float* __restrict__ hf,
                                              bf16* __restrict__ hb,
                                              float* __restrict__ ctxf) {
  int idx = blockIdx.x * 256 + threadIdx.x;
  if (idx < NB * NH) { float v = h0[idx]; hf[idx] = v; hb[idx] = (bf16)v; }
  if (idx < NB * NC) ctxf[idx] = c0[idx];
}

// ---- per-step kernels -----------------------------------------------------
__global__ __launch_bounds__(256) void k_embed(const int* __restrict__ tw_t,
                                               const float* __restrict__ emb,
                                               const float* __restrict__ ctxf,
                                               bf16* __restrict__ x) {
  int b = blockIdx.x, tid = threadIdx.x;
  int w = tw_t[b];
  const float* er = emb + (size_t)w * NE;
  for (int i = tid; i < NE; i += 256) x[b * NK1 + i] = (bf16)er[i];
  for (int i = tid; i < NC; i += 256)
    x[b * NK1 + NE + i] = (bf16)ctxf[b * NC + i];
}

// gi = x @ w_ih^T + b_ih ; gh = h @ w_hh^T + b_hh   (both 64 x 1536)
__global__ __launch_bounds__(256) void k_gru_gemm(
    const bf16* __restrict__ x, const bf16* __restrict__ hb,
    const bf16* __restrict__ wih, const bf16* __restrict__ whh,
    const float* __restrict__ bih, const float* __restrict__ bhh,
    float* __restrict__ gi, float* __restrict__ gh) {
  int lane = threadIdx.x & 31, wave = threadIdx.x >> 5;
  int tile = blockIdx.x * 8 + wave;          // 48 blocks * 8 waves = 384 tiles
  int m0 = (tile & 3) * 16;                  // M = 64  -> 4 tiles
  int n0 = (tile >> 2) * 16;                 // N = 1536 -> 96 tiles
  int ar = m0 + (lane & 15), br = n0 + (lane & 15);
  v8f ai = {}, ah = {};
  for (int k0 = 0; k0 < NK1; k0 += 32) {
    v16bf a = ld_frag(x, ar, NK1, k0, lane);
    v16bf b = ld_frag(wih, br, NK1, k0, lane);
    ai = wmma_bf16(a, b, ai);
  }
  for (int k0 = 0; k0 < NH; k0 += 32) {
    v16bf a = ld_frag(hb, ar, NH, k0, lane);
    v16bf b = ld_frag(whh, br, NH, k0, lane);
    ah = wmma_bf16(a, b, ah);
  }
  int col = lane & 15, half = lane >> 4;
  float bi = bih[n0 + col], bh = bhh[n0 + col];
#pragma unroll
  for (int r = 0; r < 8; ++r) {
    int row = m0 + r + 8 * half;             // C-layout: VGPR r -> M=r / r+8
    gi[row * NK1 + n0 + col] = ai[r] + bi;
    gh[row * NK1 + n0 + col] = ah[r] + bh;
  }
}

__global__ __launch_bounds__(256) void k_gate(const float* __restrict__ gi,
                                              const float* __restrict__ gh,
                                              float* __restrict__ hf,
                                              bf16* __restrict__ hb,
                                              bf16* __restrict__ st, int t) {
  int idx = blockIdx.x * 256 + threadIdx.x;  // 64*512
  int b = idx >> 9, n = idx & 511;
  int g = b * NK1 + n;
  float r  = 1.f / (1.f + __expf(-(gi[g] + gh[g])));
  float z  = 1.f / (1.f + __expf(-(gi[g + NH] + gh[g + NH])));
  float nn = tanhf(gi[g + 2 * NH] + r * gh[g + 2 * NH]);
  float hn = (1.f - z) * nn + z * hf[idx];
  hf[idx] = hn;
  hb[idx] = (bf16)hn;
  st[((size_t)(t * NB + b)) * NK1 + n] = (bf16)hn;   // state row: [h | ctx]
}

// q = h @ att_w  (64 x 1024), att_w pre-transposed to (C, H) bf16
__global__ __launch_bounds__(256) void k_qgemm(const bf16* __restrict__ hb,
                                               const bf16* __restrict__ awT,
                                               float* __restrict__ q) {
  int lane = threadIdx.x & 31, wave = threadIdx.x >> 5;
  int tile = blockIdx.x * 8 + wave;          // 32 blocks * 8 = 256 tiles
  int m0 = (tile & 3) * 16;
  int n0 = (tile >> 2) * 16;                 // N = 1024 -> 64 tiles
  int ar = m0 + (lane & 15), br = n0 + (lane & 15);
  v8f acc = {};
  for (int k0 = 0; k0 < NH; k0 += 32) {
    v16bf a = ld_frag(hb, ar, NH, k0, lane);
    v16bf b = ld_frag(awT, br, NH, k0, lane);
    acc = wmma_bf16(a, b, acc);
  }
  int col = lane & 15, half = lane >> 4;
#pragma unroll
  for (int r = 0; r < 8; ++r)
    q[(m0 + r + 8 * half) * NC + n0 + col] = acc[r];
}

// scores -> masked softmax -> context, one workgroup per batch element
__global__ __launch_bounds__(256) void k_attn(const float* __restrict__ q,
                                              const float* __restrict__ src,
                                              const int* __restrict__ slen,
                                              float* __restrict__ ctxf,
                                              bf16* __restrict__ st, int t) {
  __shared__ float qs[NC];
  __shared__ float sc[NS];
  __shared__ float aw[NS];
  int b = blockIdx.x, tid = threadIdx.x;
  for (int i = tid; i < NC; i += 256) qs[i] = q[b * NC + i];
  __syncthreads();
  int lane = tid & 31, wave = tid >> 5;
  for (int s = wave * 8; s < wave * 8 + 8; ++s) {      // 8 rows per wave
    const float* row = src + ((size_t)b * NS + s) * NC;
    float p = 0.f;
    for (int c = lane; c < NC; c += 32) p += row[c] * qs[c];
#pragma unroll
    for (int o = 16; o > 0; o >>= 1) p += __shfl_xor(p, o, 32);
    if (lane == 0) sc[s] = p;
  }
  __syncthreads();
  if (tid < NS) sc[tid] = (tid < slen[b]) ? sc[tid] : -1e30f;
  __syncthreads();
  if (tid < NS) {
    float mx = -3.4e38f;
    for (int j = 0; j < NS; ++j) mx = fmaxf(mx, sc[j]);
    float den = 0.f;
    for (int j = 0; j < NS; ++j) den += __expf(sc[j] - mx);
    aw[tid] = __expf(sc[tid] - mx) / den;
  }
  __syncthreads();
  for (int c = tid; c < NC; c += 256) {
    float acc = 0.f;
#pragma unroll 4
    for (int s = 0; s < NS; ++s)
      acc += aw[s] * src[((size_t)b * NS + s) * NC + c];
    ctxf[b * NC + c] = acc;
    st[((size_t)(t * NB + b)) * NK1 + NE + c] = (bf16)acc;
  }
}

// ---- phase 2: logits = state(2048x1536) @ lin_w^T(1536x32000) + b ---------
// Block tile 128(M) x 128(N); 8 waves as 4(M) x 2(N); wave tile 32x64
// (2 M-frags x 4 N-frags -> 8 WMMAs per fetch group). B tile (128x32 bf16,
// 8 KB) is double-buffered in LDS via GLOBAL_LOAD_ASYNC_TO_LDS_B128
// (ASYNCcnt), shared by the 4 M-position waves.
__global__ __launch_bounds__(256) void k_logits(const bf16* __restrict__ st,
                                                const bf16* __restrict__ lw,
                                                const float* __restrict__ lb,
                                                float* __restrict__ out) {
  __shared__ bf16 Bs[2][128 * 32];           // 2 x 8 KB
  int tid = threadIdx.x;
  int lane = tid & 31, wave = tid >> 5;
  int wm = wave & 3, wn = wave >> 2;         // wave grid: 4(M) x 2(N)
  int n_base = blockIdx.x * 128;
  int m_base = blockIdx.y * 128 + wm * 32;
  int ar0 = m_base + (lane & 15);
  int ar1 = m_base + 16 + (lane & 15);
  int half8 = (lane >> 4) << 3;

  // issue one 8 KB B-stage: 256 threads x 2 x 16B async loads (2 instr/wave)
  auto stage = [&](int buf, int k0) {
#pragma unroll
    for (int i = 0; i < 2; ++i) {
      int f = (i * 256 + tid) * 8;           // bf16 element offset in tile
      int r = f >> 5, c = f & 31;
      unsigned lds = (unsigned)(unsigned long long)(&Bs[buf][f]);
      unsigned long long ga = (unsigned long long)
          (lw + (size_t)(n_base + r) * NK1 + k0 + c);
      asm volatile("global_load_async_to_lds_b128 %0, %1, off"
                   :: "v"(lds), "v"(ga) : "memory");
    }
  };

  stage(0, 0);
  v8f acc[2][4] = {};
  for (int kt = 0; kt < NK1 / 32; ++kt) {
    int k0 = kt * 32;
    int buf = kt & 1;
    if (kt + 1 < NK1 / 32) {
      stage(buf ^ 1, k0 + 32);
      asm volatile("s_wait_asynccnt 0x2" ::: "memory");  // stage kt resident
    } else {
      asm volatile("s_wait_asynccnt 0x0" ::: "memory");
    }
    __syncthreads();
    v16bf a0 = ld_frag(st, ar0, NK1, k0, lane);
    v16bf a1 = ld_frag(st, ar1, NK1, k0, lane);
#pragma unroll
    for (int f = 0; f < 4; ++f) {
      const bf16* bp = &Bs[buf][(wn * 64 + f * 16 + (lane & 15)) * 32 + half8];
      v16bf b = combine(*(const v8bf*)bp, *(const v8bf*)(bp + 16));
      acc[0][f] = wmma_bf16(a0, b, acc[0][f]);
      acc[1][f] = wmma_bf16(a1, b, acc[1][f]);
    }
    __syncthreads();
  }

  int col = lane & 15, half = lane >> 4;
#pragma unroll
  for (int mi = 0; mi < 2; ++mi) {
#pragma unroll
    for (int f = 0; f < 4; ++f) {
      int n = n_base + wn * 64 + f * 16 + col;
      float bias = lb[n];
#pragma unroll
      for (int r = 0; r < 8; ++r) {
        int row = m_base + mi * 16 + r + 8 * half;
        out[(size_t)row * NV + n] = acc[mi][f][r] + bias;
      }
    }
  }
}

// ---------------------------------------------------------------------------
extern "C" void kernel_launch(void* const* d_in, const int* in_sizes, int n_in,
                              void* d_out, int out_size, void* d_ws,
                              size_t ws_size, hipStream_t stream) {
  const int*   slen = (const int*)d_in[0];
  const int*   tw   = (const int*)d_in[1];
  const float* src  = (const float*)d_in[2];
  const float* h0   = (const float*)d_in[3];
  const float* c0   = (const float*)d_in[4];
  const float* emb  = (const float*)d_in[5];
  const float* attw = (const float*)d_in[6];
  const float* wih  = (const float*)d_in[7];
  const float* whh  = (const float*)d_in[8];
  const float* bih  = (const float*)d_in[9];
  const float* bhh  = (const float*)d_in[10];
  const float* lw   = (const float*)d_in[11];
  const float* lb   = (const float*)d_in[12];
  float* out = (float*)d_out;
  (void)in_sizes; (void)n_in; (void)out_size; (void)ws_size;

  char* wsp = (char*)d_ws;
  auto alloc = [&](size_t bytes) {
    char* p = wsp; wsp += (bytes + 255) & ~(size_t)255; return p;
  };
  bf16*  lwb  = (bf16*)alloc((size_t)NV * NK1 * 2);    // 98.3 MB (L2-resident)
  bf16*  wihb = (bf16*)alloc((size_t)NK1 * NK1 * 2);
  bf16*  whhb = (bf16*)alloc((size_t)NK1 * NH * 2);
  bf16*  awT  = (bf16*)alloc((size_t)NC * NH * 2);
  bf16*  st   = (bf16*)alloc((size_t)NT * NB * NK1 * 2);
  bf16*  x    = (bf16*)alloc((size_t)NB * NK1 * 2);
  bf16*  hb   = (bf16*)alloc((size_t)NB * NH * 2);
  float* hf   = (float*)alloc((size_t)NB * NH * 4);
  float* ctxf = (float*)alloc((size_t)NB * NC * 4);
  float* gi   = (float*)alloc((size_t)NB * NK1 * 4);
  float* gh   = (float*)alloc((size_t)NB * NK1 * 4);
  float* q    = (float*)alloc((size_t)NB * NC * 4);

  // phase 0: one-time bf16 weight conversion
  k_cvt<<<NV * NK1 / 4 / 256, 256, 0, stream>>>(lw, lwb, NV * NK1);
  k_cvt<<<NK1 * NK1 / 4 / 256, 256, 0, stream>>>(wih, wihb, NK1 * NK1);
  k_cvt<<<NK1 * NH / 4 / 256, 256, 0, stream>>>(whh, whhb, NK1 * NH);
  k_cvt_t<<<(NH * NC + 255) / 256, 256, 0, stream>>>(attw, awT, NH, NC);
  k_init<<<(NB * NC + 255) / 256, 256, 0, stream>>>(h0, c0, hf, hb, ctxf);

  // phase 1: sequential recurrence (logits deferred)
  for (int t = 0; t < NT; ++t) {
    k_embed<<<NB, 256, 0, stream>>>(tw + t * NB, emb, ctxf, x);
    k_gru_gemm<<<48, 256, 0, stream>>>(x, hb, wihb, whhb, bih, bhh, gi, gh);
    k_gate<<<NB * NH / 256, 256, 0, stream>>>(gi, gh, hf, hb, st, t);
    k_qgemm<<<32, 256, 0, stream>>>(hb, awT, q);
    k_attn<<<NB, 256, 0, stream>>>(q, src, slen, ctxf, st, t);
  }

  // phase 2: one big WMMA GEMM for all 32 steps of logits
  k_logits<<<dim3(NV / 128, NT * NB / 128), 256, 0, stream>>>(st, lwb, lb, out);
}
